// PerViewAttention_52810917871778
// MI455X (gfx1250) — matrix operand order repaired and checked by reading, not compiled, not run on hardware
//
#include <hip/hip_runtime.h>
#include <stdint.h>

typedef __attribute__((ext_vector_type(16))) _Float16 v16h;
typedef __attribute__((ext_vector_type(4)))  _Float16 v4h;
typedef __attribute__((ext_vector_type(8)))  float    v8f;
typedef int v4i __attribute__((__vector_size__(16)));   // matches builtin proto

#define D_MODEL 1024
#define N_HEADS 8
#define D_HEAD  128
#define BATCH   8192
#define KNEI    16

#if __has_builtin(__builtin_amdgcn_global_load_async_to_lds_b128)
#define HAVE_ASYNC_LDS 1
#else
#define HAVE_ASYNC_LDS 0
#endif

// ---------------------------------------------------------------------------
// Fragment layouts (CDNA5 ISA 7.12.2, 16-bit A 16x32 / B 32x16, wave32):
//   A element (m,k): lane = 16*((k>>3)&1) + m,  slot = (k&7) + ((k>>4)<<3)
//   B element (k,n): lane = 16*((k>>3)&1) + n,  slot = (k&7) + ((k>>4)<<3)
// LDS tiles are [tile][lane][slot] halves -> one 32B contiguous read per lane.
// Weights are pre-packed to f16 in this exact order: Wp[kchunk][ntile][lane][slot]
// so the GEMM's B staging is a raw 16B copy (async global->LDS when available).
// ---------------------------------------------------------------------------

__device__ __forceinline__ v4h cvt4(float4 f) {
  v4h h;
  h[0] = (_Float16)f.x; h[1] = (_Float16)f.y;
  h[2] = (_Float16)f.z; h[3] = (_Float16)f.w;
  return h;
}

__device__ __forceinline__ void async_copy16(const _Float16* g, _Float16* l) {
#if HAVE_ASYNC_LDS
  typedef __attribute__((address_space(1))) v4i gv4i;
  typedef __attribute__((address_space(3))) v4i lv4i;
  __builtin_amdgcn_global_load_async_to_lds_b128(
      (gv4i*)(uintptr_t)g, (lv4i*)(uintptr_t)l, 0, 0);
#else
  *(uint4*)l = *(const uint4*)g;
#endif
}

__device__ __forceinline__ void wait_async0() {
#if HAVE_ASYNC_LDS
#if __has_builtin(__builtin_amdgcn_s_wait_asynccnt)
  __builtin_amdgcn_s_wait_asynccnt(0);
#else
  asm volatile("s_wait_asynccnt 0x0" ::: "memory");
#endif
#endif
}

// ---------------------------------------------------------------------------
// Weight pre-pack: W[1024,1024] f32 row-major -> f16 fragment-tiled
// Wp[kchunk(32)][ntile(64)][lane(32)][slot(16)].  Grid 1024 x 256.
// ---------------------------------------------------------------------------
__global__ __launch_bounds__(256)
void pack_weight(const float* __restrict__ W, _Float16* __restrict__ Wp) {
  int task = blockIdx.x * 256 + threadIdx.x;   // 262144 float4 tasks
  int n = task >> 8;
  int k = (task & 255) << 2;
  float4 wv = *(const float4*)&W[(size_t)n * D_MODEL + k];
  int kchunk = k >> 5, kin = k & 31;
  int lane = (((kin >> 3) & 1) << 4) | (n & 15);
  int slot = (kin & 7) + ((kin >> 4) << 3);
  size_t off = (((size_t)kchunk * 64 + (n >> 4)) * 32 + lane) * 16 + slot;
  *(v4h*)&Wp[off] = cvt4(wv);
}

// ---------------------------------------------------------------------------
// GEMM: C[M,1024] = A[M,1024] @ W^T + bias, W pre-packed f16.
// 512 threads = 16 waves (4x4). Block tile M=128, N=512, Kstep=32.
// Wave tile 32x128 -> 16 WMMAs / K-step. Double-buffered LDS, one barrier
// per K-step; next A prefetched into regs and next B issued as async
// global->LDS copies before the WMMAs so fetch overlaps compute.
// ---------------------------------------------------------------------------
template <typename OutT>
__global__ __launch_bounds__(512)
void gemm_xwt(const float* __restrict__ A, const _Float16* __restrict__ Wp,
              const float* __restrict__ bias, OutT* __restrict__ C) {
  __shared__ __align__(32) _Float16 As[2][8][32][16];   // 16 KB
  __shared__ __align__(32) _Float16 Bs[2][32][32][16];  // 64 KB

  const int tid  = threadIdx.x;
  const int lane = tid & 31;
  const int wave = tid >> 5;
  const int wm   = wave >> 2;
  const int wn   = wave & 3;
  const int mblk = blockIdx.x * 128;
  const int nblk = blockIdx.y * 512;

  v8f acc[2][8];
#pragma unroll
  for (int im = 0; im < 2; ++im)
#pragma unroll
    for (int jn = 0; jn < 8; ++jn)
#pragma unroll
      for (int i = 0; i < 8; ++i) acc[im][jn][i] = 0.0f;

  // ---- per-thread staging geometry (constant over K loop) -----------------
  // A: 1024 float4 tasks / K-step, 2 per thread.
  int a_row[2], a_j[2];
  const float* a_src[2];
  _Float16* a_dst[2][2];   // [buf][i]
#pragma unroll
  for (int i = 0; i < 2; ++i) {
    int task = tid + i * 512;
    a_row[i] = task >> 3;
    a_j[i]   = (task & 7) << 2;
    a_src[i] = &A[(size_t)(mblk + a_row[i]) * D_MODEL + a_j[i]];
    int g    = (a_j[i] >> 3) & 1;
    int slot = (a_j[i] & 7) + ((a_j[i] >> 4) << 3);
    a_dst[0][i] = &As[0][a_row[i] >> 4][(g << 4) + (a_row[i] & 15)][slot];
    a_dst[1][i] = &As[1][a_row[i] >> 4][(g << 4) + (a_row[i] & 15)][slot];
  }
  // B: 2048 16B chunks / K-step, 4 per thread; packed source is contiguous.
  const _Float16* b_src = Wp + (size_t)(nblk >> 4) * 512 + (size_t)tid * 8;
  _Float16* b_dst0 = &Bs[0][0][0][0] + (size_t)tid * 8;
  _Float16* b_dst1 = &Bs[1][0][0][0] + (size_t)tid * 8;

  // ---- prologue: prefetch K-step 0 ----------------------------------------
  float4 areg[2];
#pragma unroll
  for (int i = 0; i < 2; ++i) areg[i] = *(const float4*)(a_src[i]);
#pragma unroll
  for (int c = 0; c < 4; ++c)
    async_copy16(b_src + (size_t)c * 4096, b_dst0 + (size_t)c * 4096);

  for (int it = 0; it < 32; ++it) {
    const int p = it & 1;
    // commit prefetched A for this step (f32 -> f16 fragment layout)
#pragma unroll
    for (int i = 0; i < 2; ++i) *(v4h*)a_dst[p][i] = cvt4(areg[i]);
    wait_async0();
    __syncthreads();

    // issue next K-step fetches (overlap with WMMAs below)
    if (it + 1 < 32) {
      const int kk = (it + 1) * 32;
#pragma unroll
      for (int i = 0; i < 2; ++i) areg[i] = *(const float4*)(a_src[i] + kk);
      const _Float16* bs = b_src + (size_t)(it + 1) * 32768;  // 64 ntiles*512
      _Float16* bd = p ? b_dst0 : b_dst1;
#pragma unroll
      for (int c = 0; c < 4; ++c)
        async_copy16(bs + (size_t)c * 4096, bd + (size_t)c * 4096);
    }

    // 16 WMMAs on buffer p
    v16h a0 = *(const v16h*)&As[p][wm * 2 + 0][lane][0];
    v16h a1 = *(const v16h*)&As[p][wm * 2 + 1][lane][0];
#pragma unroll
    for (int jn = 0; jn < 8; ++jn) {
      v16h b = *(const v16h*)&Bs[p][wn * 8 + jn][lane][0];
      acc[0][jn] = __builtin_amdgcn_wmma_f32_16x16x32_f16(
          false, a0, false, b, (short)0, acc[0][jn], false, false);
      acc[1][jn] = __builtin_amdgcn_wmma_f32_16x16x32_f16(
          false, a1, false, b, (short)0, acc[1][jn], false, false);
    }
  }

  // ---- epilogue: C reg r -> (m = r + 8*(lane/16), n = lane%16) ------------
  const int g  = lane >> 4;
  const int ln = lane & 15;
#pragma unroll
  for (int im = 0; im < 2; ++im) {
    int mrow = mblk + wm * 32 + im * 16 + (g << 3);
#pragma unroll
    for (int jn = 0; jn < 8; ++jn) {
      int n = nblk + wn * 128 + jn * 16 + ln;
      float bv = bias[n];
#pragma unroll
      for (int r = 0; r < 8; ++r)
        C[(size_t)(mrow + r) * D_MODEL + n] = (OutT)(acc[im][jn][r] + bv);
    }
  }
}

// ---------------------------------------------------------------------------
// Single-query attention: one wave per (b, h). 4 dims per lane (Dh=128).
// ---------------------------------------------------------------------------
__global__ __launch_bounds__(256)
void attention_kernel(const float* __restrict__ Q, const _Float16* __restrict__ Kh,
                      const _Float16* __restrict__ Vh, const float* __restrict__ log_ew,
                      float* __restrict__ ctx) {
  const int lane = threadIdx.x & 31;
  const int wave = threadIdx.x >> 5;
  const int task = blockIdx.x * 8 + wave;   // b*H + h
  const int b = task >> 3;
  const int h = task & 7;
  const int d0 = lane << 2;
  const size_t qoff = (size_t)b * D_MODEL + h * D_HEAD + d0;
  const float4 q = *(const float4*)&Q[qoff];

  float s[KNEI];
#pragma unroll
  for (int k = 0; k < KNEI; ++k) {
    const v4h kv = *(const v4h*)&Kh[(size_t)(b * KNEI + k) * D_MODEL + h * D_HEAD + d0];
    float p = q.x * (float)kv[0] + q.y * (float)kv[1] +
              q.z * (float)kv[2] + q.w * (float)kv[3];
#pragma unroll
    for (int off = 16; off > 0; off >>= 1) p += __shfl_xor(p, off, 32);
    s[k] = p * 0.08838834764831845f + log_ew[b * KNEI + k];  // 1/sqrt(128)
  }
  float mx = s[0];
#pragma unroll
  for (int k = 1; k < KNEI; ++k) mx = fmaxf(mx, s[k]);
  float sum = 0.0f;
#pragma unroll
  for (int k = 0; k < KNEI; ++k) { s[k] = __expf(s[k] - mx); sum += s[k]; }
  const float inv = 1.0f / sum;

  float4 c = {0.0f, 0.0f, 0.0f, 0.0f};
#pragma unroll
  for (int k = 0; k < KNEI; ++k) {
    const v4h vv = *(const v4h*)&Vh[(size_t)(b * KNEI + k) * D_MODEL + h * D_HEAD + d0];
    float a = s[k] * inv;
    c.x += a * (float)vv[0]; c.y += a * (float)vv[1];
    c.z += a * (float)vv[2]; c.w += a * (float)vv[3];
  }
  *(float4*)&ctx[qoff] = c;
}

// ---------------------------------------------------------------------------
// out = ctx @ Wo^T + bo with fused LayerNorm. 256 threads = 8 waves (1x8).
// Block tile M=16, N=1024 (full row in-block -> LN via ds_add_f32 reductions).
// Wo pre-packed f16; B staging is raw (async) copy.
// ---------------------------------------------------------------------------
__global__ __launch_bounds__(256)
void out_ln_kernel(const float* __restrict__ X, const _Float16* __restrict__ Wop,
                   const float* __restrict__ bo, const float* __restrict__ gamma,
                   const float* __restrict__ beta, float* __restrict__ out) {
  __shared__ __align__(32) _Float16 As[32][16];        // 1 KB
  __shared__ __align__(32) _Float16 Bs[64][32][16];    // 64 KB
  __shared__ float red[2][16];

  const int tid  = threadIdx.x;
  const int lane = tid & 31;
  const int wave = tid >> 5;          // owns n in [wave*128, wave*128+128)
  const int mblk = blockIdx.x * 16;

  if (tid < 32) red[tid >> 4][tid & 15] = 0.0f;

  v8f acc[8];
#pragma unroll
  for (int t = 0; t < 8; ++t)
#pragma unroll
    for (int i = 0; i < 8; ++i) acc[t][i] = 0.0f;

  const _Float16* b_src = Wop + (size_t)tid * 8;       // packed, contiguous
  _Float16* b_dst = &Bs[0][0][0] + (size_t)tid * 8;

  for (int kk = 0; kk < D_MODEL; kk += 32) {
    if (tid < 128) {                      // A: 16 rows x 32 k, f32 -> f16
      int row = tid >> 3, j = (tid & 7) << 2;
      float4 av = *(const float4*)&X[(size_t)(mblk + row) * D_MODEL + kk + j];
      int g = (j >> 3) & 1, slot = (j & 7) + ((j >> 4) << 3);
      *(v4h*)&As[(g << 4) + (row & 15)][slot] = cvt4(av);
    }
    // B: 64 ntiles x 512 halves (64 KB), 16 x 16B chunks per thread
    const _Float16* bs = b_src + (size_t)(kk >> 5) * 32768;
#pragma unroll
    for (int c = 0; c < 16; ++c)
      async_copy16(bs + (size_t)c * 2048, b_dst + (size_t)c * 2048);
    wait_async0();
    __syncthreads();

    v16h a = *(const v16h*)&As[lane][0];
#pragma unroll
    for (int t = 0; t < 8; ++t) {
      v16h b = *(const v16h*)&Bs[wave * 8 + t][lane][0];
      acc[t] = __builtin_amdgcn_wmma_f32_16x16x32_f16(
          false, a, false, b, (short)0, acc[t], false, false);
    }
    __syncthreads();
  }

  // ---- fused bias + LayerNorm ---------------------------------------------
  const int g = lane >> 4, ln = lane & 15;
  const int rowb = g << 3;
  int   ncol[8];
  float gn[8], bn[8];
#pragma unroll
  for (int t = 0; t < 8; ++t) {
    int n = wave * 128 + t * 16 + ln;
    ncol[t] = n;
    float bval = bo[n];
    gn[t] = gamma[n];
    bn[t] = beta[n];
#pragma unroll
    for (int r = 0; r < 8; ++r) acc[t][r] += bval;
  }
#pragma unroll
  for (int r = 0; r < 8; ++r) {
    float su = 0.0f, ss = 0.0f;
#pragma unroll
    for (int t = 0; t < 8; ++t) { float v = acc[t][r]; su += v; ss += v * v; }
    atomicAdd(&red[0][rowb + r], su);    // ds_add_f32
    atomicAdd(&red[1][rowb + r], ss);
  }
  __syncthreads();
#pragma unroll
  for (int r = 0; r < 8; ++r) {
    float mu   = red[0][rowb + r] * (1.0f / 1024.0f);
    float var  = red[1][rowb + r] * (1.0f / 1024.0f) - mu * mu;
    float rstd = rsqrtf(var + 1e-5f);
    size_t rowoff = (size_t)(mblk + rowb + r) * D_MODEL;
#pragma unroll
    for (int t = 0; t < 8; ++t)
      out[rowoff + ncol[t]] = (acc[t][r] - mu) * rstd * gn[t] + bn[t];
  }
}

// ---------------------------------------------------------------------------
extern "C" void kernel_launch(void* const* d_in, const int* in_sizes, int n_in,
                              void* d_out, int out_size, void* d_ws, size_t ws_size,
                              hipStream_t stream) {
  (void)in_sizes; (void)n_in; (void)out_size; (void)ws_size;
  const float* z_anc  = (const float*)d_in[0];
  const float* z_nei  = (const float*)d_in[1];
  const float* v_aug  = (const float*)d_in[2];
  const float* log_ew = (const float*)d_in[3];
  const float* Wq = (const float*)d_in[4];
  const float* bq = (const float*)d_in[5];
  const float* Wk = (const float*)d_in[6];
  const float* bk = (const float*)d_in[7];
  const float* Wv = (const float*)d_in[8];
  const float* bv = (const float*)d_in[9];
  const float* Wo = (const float*)d_in[10];
  const float* bo = (const float*)d_in[11];
  const float* gamma = (const float*)d_in[12];
  const float* beta  = (const float*)d_in[13];
  float* out = (float*)d_out;

  // Workspace (584 MB): Q f32 32MB | Kh f16 256MB | Vh f16 256MB |
  //                     ctx f32 32MB | 4x packed weights f16 2MB each
  char* ws = (char*)d_ws;
  float*    Qws   = (float*)ws;     ws += (size_t)BATCH * D_MODEL * sizeof(float);
  _Float16* Khws  = (_Float16*)ws;  ws += (size_t)BATCH * KNEI * D_MODEL * sizeof(_Float16);
  _Float16* Vhws  = (_Float16*)ws;  ws += (size_t)BATCH * KNEI * D_MODEL * sizeof(_Float16);
  float*    ctxws = (float*)ws;     ws += (size_t)BATCH * D_MODEL * sizeof(float);
  _Float16* Wqp = (_Float16*)ws;    ws += (size_t)D_MODEL * D_MODEL * sizeof(_Float16);
  _Float16* Wkp = (_Float16*)ws;    ws += (size_t)D_MODEL * D_MODEL * sizeof(_Float16);
  _Float16* Wvp = (_Float16*)ws;    ws += (size_t)D_MODEL * D_MODEL * sizeof(_Float16);
  _Float16* Wop = (_Float16*)ws;

  const dim3 blk512(512), blk256(256);
  pack_weight<<<dim3(1024), blk256, 0, stream>>>(Wq, Wqp);
  pack_weight<<<dim3(1024), blk256, 0, stream>>>(Wk, Wkp);
  pack_weight<<<dim3(1024), blk256, 0, stream>>>(Wv, Wvp);
  pack_weight<<<dim3(1024), blk256, 0, stream>>>(Wo, Wop);

  gemm_xwt<float><<<dim3(BATCH / 128, D_MODEL / 512), blk512, 0, stream>>>(
      z_anc, Wqp, bq, Qws);
  gemm_xwt<_Float16><<<dim3((BATCH * KNEI) / 128, D_MODEL / 512), blk512, 0, stream>>>(
      z_nei, Wkp, bk, Khws);
  gemm_xwt<_Float16><<<dim3((BATCH * KNEI) / 128, D_MODEL / 512), blk512, 0, stream>>>(
      v_aug, Wvp, bv, Vhws);
  attention_kernel<<<dim3((BATCH * N_HEADS) / 8), blk256, 0, stream>>>(
      Qws, Khws, Vhws, log_ew, ctxws);
  out_ln_kernel<<<dim3(BATCH / 16), blk256, 0, stream>>>(
      ctxws, Wop, bo, gamma, beta, out);
}
